// EvolveGCNClassifier_15144054685719
// MI455X (gfx1250) — compile-verified, matrix operand order checked by reading, not run on hardware
//
#include <hip/hip_runtime.h>
#include <hip/hip_bf16.h>
#include <math.h>

#define N_NODES 200000
#define N_EDGES 500000
#define D_IN    165
#define D_HID   128
#define D_OUT   2
#define KB      6      // ceil(165/32) K-blocks of 32
#define NT1     11     // ceil(165/16) N tiles for GEMM1
#define NT2     8      // 128/16      N tiles for GEMM2

typedef __attribute__((ext_vector_type(16))) __bf16 v16bf;
typedef __attribute__((ext_vector_type(8)))  float  v8f;
typedef int v4i __attribute__((vector_size(16)));

struct Frag32B { uint4 lo, hi; };

static __device__ __forceinline__ v16bf make_frag(uint4 lo, uint4 hi) {
  Frag32B t{lo, hi};
  return __builtin_bit_cast(v16bf, t);
}

// float -> bf16 round-to-nearest-even (bit trick; NaN edge ignored for this data)
static __device__ __forceinline__ unsigned short f2bf(float f) {
  unsigned int u = __builtin_bit_cast(unsigned int, f);
  u += 0x7FFFu + ((u >> 16) & 1u);
  return (unsigned short)(u >> 16);
}

static __device__ __forceinline__ unsigned int pack_bf2(float a, float b) {
  return (unsigned int)f2bf(a) | ((unsigned int)f2bf(b) << 16);
}

// 16-byte global -> LDS copy; async global->LDS DMA path when available.
static __device__ __forceinline__ void copy16_g2l(unsigned short* l,
                                                  const unsigned short* g) {
#if __has_builtin(__builtin_amdgcn_global_load_async_to_lds_b128)
  __builtin_amdgcn_global_load_async_to_lds_b128(
      (__attribute__((address_space(1))) v4i*)g,
      (__attribute__((address_space(3))) v4i*)l, 0, 0);
#else
  *(uint4*)l = *(const uint4*)g;
#endif
}

static __device__ __forceinline__ void wait_async0() {
#if __has_builtin(__builtin_amdgcn_s_wait_asynccnt)
  __builtin_amdgcn_s_wait_asynccnt(0);
#else
  asm volatile("s_wait_asynccnt 0x0" ::: "memory");
#endif
}

// ---------------------------------------------------------------------------
// 1) GRU weight evolution: W_t = GRU(W)  (165x165, tiny)
// ---------------------------------------------------------------------------
__global__ void gru_evolve_kernel(const float* __restrict__ W,
                                  const float* __restrict__ Wih,
                                  const float* __restrict__ Whh,
                                  const float* __restrict__ bih,
                                  const float* __restrict__ bhh,
                                  float* __restrict__ Wt) {
  __shared__ float wrow[D_IN];
  const int i = blockIdx.x;
  for (int k = threadIdx.x; k < D_IN; k += blockDim.x) wrow[k] = W[i * D_IN + k];
  __syncthreads();
  const int j = threadIdx.x;
  if (j >= D_IN) return;
  float g0 = bih[j], g1 = bih[j + D_IN], g2 = bih[j + 2 * D_IN];
  float g3 = bhh[j], g4 = bhh[j + D_IN], g5 = bhh[j + 2 * D_IN];
  for (int k = 0; k < D_IN; ++k) {
    float w = wrow[k];
    g0 += w * Wih[(size_t)(j            ) * D_IN + k];
    g1 += w * Wih[(size_t)(j +     D_IN ) * D_IN + k];
    g2 += w * Wih[(size_t)(j + 2 * D_IN ) * D_IN + k];
    g3 += w * Whh[(size_t)(j            ) * D_IN + k];
    g4 += w * Whh[(size_t)(j +     D_IN ) * D_IN + k];
    g5 += w * Whh[(size_t)(j + 2 * D_IN ) * D_IN + k];
  }
  float r  = 1.0f / (1.0f + expf(-(g0 + g3)));
  float z  = 1.0f / (1.0f + expf(-(g1 + g4)));
  float nn = tanhf(g2 + r * g5);
  Wt[i * D_IN + j] = (1.0f - z) * nn + z * W[i * D_IN + j];
}

// ---------------------------------------------------------------------------
// 2) Pre-swizzle B matrices into WMMA fragment order (bf16).
//    Fragment element e of lane (half=lane>>4, idx=lane&15):
//      k = kb*32 + (e<8 ? half*8+e : 16+half*8+(e-8)),  n = nt*16+idx
//    B1[k][n] = W_t[k][n]            (for x1 = x @ W_t)
//    B2[k][n] = proj_W[n][k]         (for h = A @ proj_W^T)
// ---------------------------------------------------------------------------
__global__ void pack_b_kernel(const float* __restrict__ Wt,
                              const float* __restrict__ projW,
                              unsigned short* __restrict__ b1p,
                              unsigned short* __restrict__ b2p) {
  const int f    = blockIdx.x;
  const int lane = threadIdx.x;
  const int half = lane >> 4, idx = lane & 15;
  const bool isB1 = (f < KB * NT1);
  int kb, nt;
  unsigned short* dst;
  if (isB1) { kb = f / NT1; nt = f % NT1; dst = b1p + ((size_t)f * 32 + lane) * 16; }
  else      { int f2 = f - KB * NT1; kb = f2 / NT2; nt = f2 % NT2;
              dst = b2p + ((size_t)f2 * 32 + lane) * 16; }
#pragma unroll
  for (int e = 0; e < 16; ++e) {
    int k = kb * 32 + ((e < 8) ? (half * 8 + e) : (16 + half * 8 + (e - 8)));
    int n = nt * 16 + idx;
    float v = 0.0f;
    if (isB1) { if (k < D_IN && n < D_IN ) v = Wt[(size_t)k * D_IN + n]; }
    else      { if (k < D_IN && n < D_HID) v = projW[(size_t)n * D_IN + k]; }
    dst[e] = f2bf(v);
  }
}

// ---------------------------------------------------------------------------
// 3) init: agg = 0, deg = self-loop weight (2.0)
// ---------------------------------------------------------------------------
__global__ void init_kernel(float* __restrict__ agg, float* __restrict__ deg) {
  size_t i = (size_t)blockIdx.x * blockDim.x + threadIdx.x;
  const size_t total = (size_t)N_NODES * D_IN;
  if (i < total)   agg[i] = 0.0f;
  if (i < N_NODES) deg[i] = 2.0f;
}

__global__ void deg_kernel(const long long* __restrict__ ei, float* __restrict__ deg) {
  int e = blockIdx.x * blockDim.x + threadIdx.x;
  if (e < N_EDGES) atomicAdd(&deg[(int)ei[N_EDGES + e]], 1.0f);
}

__global__ void dinv_kernel(const float* __restrict__ deg, float* __restrict__ dinv) {
  int i = blockIdx.x * blockDim.x + threadIdx.x;
  if (i < N_NODES) dinv[i] = rsqrtf(deg[i]);
}

// ---------------------------------------------------------------------------
// 4) GEMM1: x1 = x @ W_t   (M=200000, K=165->192, N=165->176), bf16 WMMA
//    Block: 128 threads = 4 waves; each wave owns a 16-row strip; block = 64 rows.
//    B fragments are pre-swizzled in global (L2-hot), A staged/converted in LDS.
// ---------------------------------------------------------------------------
__global__ void __launch_bounds__(128) gemm1_x_wt(const float* __restrict__ x,
                                                  const unsigned short* __restrict__ b1p,
                                                  float* __restrict__ x1) {
  __shared__ unsigned short At[64 * 32];   // 64 rows x 32 k, bf16 (4 KB)
  const int tid  = threadIdx.x;
  const int wave = tid >> 5, lane = tid & 31;
  const int half = lane >> 4, idx = lane & 15;
  const int m_base = blockIdx.x * 64;
  v8f acc[NT1] = {};
  const uint4* bv = (const uint4*)b1p;

  for (int kb = 0; kb < KB; ++kb) {
    __syncthreads();
    {
      const int   m_loc = tid >> 1;
      const int   kseg  = (tid & 1) * 16;
      const int   gk0   = kb * 32 + kseg;
      const float* srcrow = x + (size_t)(m_base + m_loc) * D_IN;
      float v[16];
      if (gk0 + 16 <= D_IN) {            // fully in range (all but last K-block)
#pragma unroll
        for (int q = 0; q < 4; ++q) {
          float4 t;
          __builtin_memcpy(&t, srcrow + gk0 + 4 * q, sizeof(t));
          v[4 * q + 0] = t.x; v[4 * q + 1] = t.y;
          v[4 * q + 2] = t.z; v[4 * q + 3] = t.w;
        }
      } else {
#pragma unroll
        for (int j = 0; j < 16; ++j) {
          int gk = gk0 + j;
          v[j] = (gk < D_IN) ? srcrow[gk] : 0.0f;
        }
      }
      unsigned int pk[8];
#pragma unroll
      for (int q = 0; q < 8; ++q) pk[q] = pack_bf2(v[2 * q], v[2 * q + 1]);
      uint4* dstl = (uint4*)&At[m_loc * 32 + kseg];
      dstl[0] = make_uint4(pk[0], pk[1], pk[2], pk[3]);
      dstl[1] = make_uint4(pk[4], pk[5], pk[6], pk[7]);
      if (kb + 1 < KB) __builtin_prefetch(srcrow + gk0 + 32, 0, 1);  // next K slab
    }
    __syncthreads();

    const uint4* rp = (const uint4*)&At[(wave * 16 + idx) * 32];
    v16bf fa = make_frag(rp[half], rp[2 + half]);
#pragma unroll
    for (int nt = 0; nt < NT1; ++nt) {
      int fi = ((kb * NT1 + nt) * 32 + lane) * 2;
      v16bf fb = make_frag(bv[fi], bv[fi + 1]);
      acc[nt] = __builtin_amdgcn_wmma_f32_16x16x32_bf16(
          false, fa, false, fb, (short)0, acc[nt], false, false);
    }
  }

  // C layout: lanes 0-15 -> N=lane, M=r; lanes 16-31 -> N=lane-16, M=8+r
  const int mrow0 = m_base + wave * 16 + half * 8;
#pragma unroll
  for (int nt = 0; nt < NT1; ++nt) {
    int n = nt * 16 + idx;
    if (n < D_IN) {
#pragma unroll
      for (int r = 0; r < 8; ++r)
        x1[(size_t)(mrow0 + r) * D_IN + n] = acc[nt][r];
    }
  }
}

// ---------------------------------------------------------------------------
// 5) Edge scatter: agg[dst] += dinv[src]*dinv[dst] * x1[src]  (one wave/edge)
// ---------------------------------------------------------------------------
__global__ void edge_scatter_kernel(const long long* __restrict__ ei,
                                    const float* __restrict__ dinv,
                                    const float* __restrict__ x1,
                                    float* __restrict__ agg) {
  int e    = blockIdx.x * 8 + (threadIdx.x >> 5);
  int lane = threadIdx.x & 31;
  if (e >= N_EDGES) return;
  int s = (int)ei[e];
  int d = (int)ei[N_EDGES + e];
  float nm = dinv[s] * dinv[d];
  const float* xs = x1 + (size_t)s * D_IN;
  float*       ad = agg + (size_t)d * D_IN;
  for (int c = lane; c < D_IN; c += 32)
    atomicAdd(&ad[c], nm * xs[c]);
}

// ---------------------------------------------------------------------------
// 6) GEMM2 fused: A = agg + 2*dinv^2*x1 + gcn_bias (built in LDS as bf16)
//    h = relu(A @ proj_W^T + proj_b); out = h @ cls_W^T + cls_b
//    B2 (48 KB, fragment-swizzled) is staged whole into LDS via async
//    global->LDS DMA; classifier GEMM folded into the epilogue (h never
//    touches memory).
// ---------------------------------------------------------------------------
__global__ void __launch_bounds__(128) gemm2_fused(const float* __restrict__ agg,
                                                   const float* __restrict__ x1,
                                                   const float* __restrict__ dinv,
                                                   const float* __restrict__ gcn_bias,
                                                   const unsigned short* __restrict__ b2p,
                                                   const float* __restrict__ proj_b,
                                                   const float* __restrict__ cls_W,
                                                   const float* __restrict__ cls_b,
                                                   float* __restrict__ out) {
  __shared__ unsigned short Bl[KB * NT2 * 32 * 16];  // 48 KB packed B2
  __shared__ unsigned short At[64 * 32];             // 4 KB A tile
  const int tid  = threadIdx.x;
  const int wave = tid >> 5, lane = tid & 31;
  const int half = lane >> 4, idx = lane & 15;
  const int m_base = blockIdx.x * 64;
  v8f acc[NT2] = {};

  // Stage all B fragments into LDS (async DMA when available).
  {
    const int total_u4 = (KB * NT2 * 32 * 16) / 8;   // 3072 x 16B
    for (int t = tid; t < total_u4; t += 128)
      copy16_g2l(&Bl[t * 8], &b2p[t * 8]);
    wait_async0();
    __syncthreads();
  }

  for (int kb = 0; kb < KB; ++kb) {
    __syncthreads();
    {
      const int m_loc = tid >> 1;
      const int kseg  = (tid & 1) * 16;
      const int gk0   = kb * 32 + kseg;
      const size_t row = (size_t)(m_base + m_loc) * D_IN;
      const float di = dinv[m_base + m_loc];
      const float sc = 2.0f * di * di;               // self_loop_w * dinv^2
      float v[16];
      if (gk0 + 16 <= D_IN) {
#pragma unroll
        for (int q = 0; q < 4; ++q) {
          float4 ta, tx, tb;
          __builtin_memcpy(&ta, agg + row + gk0 + 4 * q, sizeof(ta));
          __builtin_memcpy(&tx, x1  + row + gk0 + 4 * q, sizeof(tx));
          __builtin_memcpy(&tb, gcn_bias + gk0 + 4 * q, sizeof(tb));
          v[4 * q + 0] = ta.x + sc * tx.x + tb.x;
          v[4 * q + 1] = ta.y + sc * tx.y + tb.y;
          v[4 * q + 2] = ta.z + sc * tx.z + tb.z;
          v[4 * q + 3] = ta.w + sc * tx.w + tb.w;
        }
      } else {
#pragma unroll
        for (int j = 0; j < 16; ++j) {
          int gk = gk0 + j;
          v[j] = (gk < D_IN) ? (agg[row + gk] + sc * x1[row + gk] + gcn_bias[gk])
                             : 0.0f;
        }
      }
      unsigned int pk[8];
#pragma unroll
      for (int q = 0; q < 8; ++q) pk[q] = pack_bf2(v[2 * q], v[2 * q + 1]);
      uint4* dstl = (uint4*)&At[m_loc * 32 + kseg];
      dstl[0] = make_uint4(pk[0], pk[1], pk[2], pk[3]);
      dstl[1] = make_uint4(pk[4], pk[5], pk[6], pk[7]);
      if (kb + 1 < KB) {
        __builtin_prefetch(agg + row + gk0 + 32, 0, 1);
        __builtin_prefetch(x1  + row + gk0 + 32, 0, 1);
      }
    }
    __syncthreads();

    const uint4* rp = (const uint4*)&At[(wave * 16 + idx) * 32];
    v16bf fa = make_frag(rp[half], rp[2 + half]);
#pragma unroll
    for (int nt = 0; nt < NT2; ++nt) {
      const uint4* blp = (const uint4*)&Bl[((kb * NT2 + nt) * 32 + lane) * 16];
      v16bf fb = make_frag(blp[0], blp[1]);
      acc[nt] = __builtin_amdgcn_wmma_f32_16x16x32_bf16(
          false, fa, false, fb, (short)0, acc[nt], false, false);
    }
  }

  // Epilogue: relu(+proj_b), then fold tiny classifier GEMM (N=2).
  float p0[8] = {}, p1[8] = {};
#pragma unroll
  for (int nt = 0; nt < NT2; ++nt) {
    int n = nt * 16 + idx;                  // n < 128 always
    float pb = proj_b[n];
    float c0 = cls_W[n];
    float c1 = cls_W[D_HID + n];
#pragma unroll
    for (int r = 0; r < 8; ++r) {
      float h = acc[nt][r] + pb;
      h = h > 0.0f ? h : 0.0f;
      p0[r] += h * c0;
      p1[r] += h * c1;
    }
  }
  // Reduce across the 16 lanes holding the same row (xor within half).
#pragma unroll
  for (int off = 8; off > 0; off >>= 1) {
#pragma unroll
    for (int r = 0; r < 8; ++r) {
      p0[r] += __shfl_xor(p0[r], off, 32);
      p1[r] += __shfl_xor(p1[r], off, 32);
    }
  }
  if (idx == 0) {
    const int m0 = m_base + wave * 16 + half * 8;
    const float cb0 = cls_b[0], cb1 = cls_b[1];
#pragma unroll
    for (int r = 0; r < 8; ++r) {
      out[(size_t)(m0 + r) * 2 + 0] = p0[r] + cb0;
      out[(size_t)(m0 + r) * 2 + 1] = p1[r] + cb1;
    }
  }
}

// ---------------------------------------------------------------------------
extern "C" void kernel_launch(void* const* d_in, const int* in_sizes, int n_in,
                              void* d_out, int out_size, void* d_ws, size_t ws_size,
                              hipStream_t stream) {
  const float*     x     = (const float*)d_in[0];
  const long long* ei    = (const long long*)d_in[1];   // int64 per reference
  const float*     W     = (const float*)d_in[2];
  const float*     Wih   = (const float*)d_in[3];
  const float*     Whh   = (const float*)d_in[4];
  const float*     bih   = (const float*)d_in[5];
  const float*     bhh   = (const float*)d_in[6];
  const float*     gbias = (const float*)d_in[7];
  const float*     projW = (const float*)d_in[8];
  const float*     projb = (const float*)d_in[9];
  const float*     clsW  = (const float*)d_in[10];
  const float*     clsb  = (const float*)d_in[11];
  float*           out   = (float*)d_out;

  char* ws = (char*)d_ws;
  auto alloc = [&](size_t bytes) -> char* {
    char* p = ws;
    ws += (bytes + 255) & ~(size_t)255;
    return p;
  };
  float*          Wt   = (float*)alloc((size_t)D_IN * D_IN * sizeof(float));
  unsigned short* b1p  = (unsigned short*)alloc((size_t)KB * NT1 * 32 * 16 * 2);
  unsigned short* b2p  = (unsigned short*)alloc((size_t)KB * NT2 * 32 * 16 * 2);
  float*          deg  = (float*)alloc((size_t)N_NODES * sizeof(float));
  float*          dnv  = (float*)alloc((size_t)N_NODES * sizeof(float));
  float*          x1   = (float*)alloc((size_t)N_NODES * D_IN * sizeof(float));
  float*          agg  = (float*)alloc((size_t)N_NODES * D_IN * sizeof(float));

  gru_evolve_kernel<<<D_IN, 192, 0, stream>>>(W, Wih, Whh, bih, bhh, Wt);
  pack_b_kernel<<<KB * NT1 + KB * NT2, 32, 0, stream>>>(Wt, projW, b1p, b2p);

  {
    size_t total = (size_t)N_NODES * D_IN;
    init_kernel<<<(unsigned)((total + 255) / 256), 256, 0, stream>>>(agg, deg);
  }
  deg_kernel<<<(N_EDGES + 255) / 256, 256, 0, stream>>>(ei, deg);
  dinv_kernel<<<(N_NODES + 255) / 256, 256, 0, stream>>>(deg, dnv);

  gemm1_x_wt<<<N_NODES / 64, 128, 0, stream>>>(x, b1p, x1);
  edge_scatter_kernel<<<(N_EDGES + 7) / 8, 256, 0, stream>>>(ei, dnv, x1, agg);
  gemm2_fused<<<N_NODES / 64, 128, 0, stream>>>(agg, x1, dnv, gbias, b2p,
                                                projb, clsW, clsb, out);
}
